// QuantizedLinear_2173253452596
// MI455X (gfx1250) — matrix operand order repaired and checked by reading, not compiled, not run on hardware
//
#include <hip/hip_runtime.h>

// QuantizedLinear for MI455X (gfx1250, wave32, WMMA, async-to-LDS).
// y = x @ (W_int8 * scale[o]).T + bias     M=8192, N=16384, K=4096
//
//  1) one-shot convert x (f32) -> f16 into ws[0 .. 64MB)
//  2) one-shot convert W (int32 holding int8 values, exact in f16) -> f16
//     into ws[64MB .. 192MB)              (requires ws_size >= 192 MB)
//  3) tiled f16 WMMA GEMM, f32 accumulation; per-channel scale + bias in the
//     f32 epilogue (weight path numerically exact).
//     Block 128x256, 8 wave32 as 2(M)x4(N), wave tile 64x64 = 4x4 WMMA tiles.
//     Global->LDS staging uses GLOBAL_LOAD_ASYNC_TO_LDS_B128 (ASYNCcnt) with
//     double-buffered dynamic LDS (110,592 B < 320 KB WGP LDS).

typedef __attribute__((ext_vector_type(16))) _Float16 v16h;
typedef __attribute__((ext_vector_type(8)))  _Float16 v8h;
typedef __attribute__((ext_vector_type(8)))  float    v8f;
typedef __attribute__((ext_vector_type(4)))  float    v4f;
typedef __attribute__((ext_vector_type(4)))  int      v4i;

union F16Frag { v16h v; v8h h[2]; };

#define B_DIM   4
#define S_DIM   2048
#define IN_DIM  4096
#define OUT_DIM 16384
#define M_TOTAL (B_DIM * S_DIM)      // 8192

#define BM 128
#define BN 256
#define BK 64
#define LDT (BK + 8)                            // padded LDS stride (halfs), 144B = 36 banks
#define A_BYTES (BM * LDT * 2)                  // 18,432
#define B_BYTES (BN * LDT * 2)                  // 36,864
#define BUF_BYTES (A_BYTES + B_BYTES)           // 55,296
#define LDS_TOTAL (2 * BUF_BYTES)               // 110,592

// ---------------- conversion kernels ----------------

__global__ void __launch_bounds__(256)
qlin_cvt_x_f16(const float* __restrict__ src, _Float16* __restrict__ dst) {
    size_t i = ((size_t)blockIdx.x * blockDim.x + threadIdx.x) * 8;
    v4f a = *(const v4f*)(src + i);
    v4f b = *(const v4f*)(src + i + 4);
    v8h o;
    o[0] = (_Float16)a[0]; o[1] = (_Float16)a[1];
    o[2] = (_Float16)a[2]; o[3] = (_Float16)a[3];
    o[4] = (_Float16)b[0]; o[5] = (_Float16)b[1];
    o[6] = (_Float16)b[2]; o[7] = (_Float16)b[3];
    *(v8h*)(dst + i) = o;
}

__global__ void __launch_bounds__(256)
qlin_cvt_w_f16(const int* __restrict__ src, _Float16* __restrict__ dst) {
    size_t i = ((size_t)blockIdx.x * blockDim.x + threadIdx.x) * 8;
    v4i a = *(const v4i*)(src + i);
    v4i b = *(const v4i*)(src + i + 4);
    v8h o;                                      // |w| <= 128: exact in f16
    o[0] = (_Float16)a[0]; o[1] = (_Float16)a[1];
    o[2] = (_Float16)a[2]; o[3] = (_Float16)a[3];
    o[4] = (_Float16)b[0]; o[5] = (_Float16)b[1];
    o[6] = (_Float16)b[2]; o[7] = (_Float16)b[3];
    *(v8h*)(dst + i) = o;
}

// ---------------- async copy helpers (CDNA5 GLOBAL_LOAD_ASYNC_TO_LDS) -------
// ISA 10.x / 15.18.3: copies global memory directly into LDS; tracked by ASYNCcnt.
// Low 32 bits of a flat pointer into LDS are the LDS byte address (ISA 10.2).

__device__ __forceinline__ unsigned lds_addr(const void* p) {
    return (unsigned)(size_t)p;
}

__device__ __forceinline__ void async_copy_64B(unsigned lds, const _Float16* g) {
    asm volatile(
        "global_load_async_to_lds_b128 %0, %1, off\n\t"
        "global_load_async_to_lds_b128 %0, %1, off offset:16\n\t"
        "global_load_async_to_lds_b128 %0, %1, off offset:32\n\t"
        "global_load_async_to_lds_b128 %0, %1, off offset:48"
        :: "v"(lds), "v"(g) : "memory");
}

__device__ __forceinline__ void async_copy_128B(unsigned lds, const _Float16* g) {
    asm volatile(
        "global_load_async_to_lds_b128 %0, %1, off\n\t"
        "global_load_async_to_lds_b128 %0, %1, off offset:16\n\t"
        "global_load_async_to_lds_b128 %0, %1, off offset:32\n\t"
        "global_load_async_to_lds_b128 %0, %1, off offset:48\n\t"
        "global_load_async_to_lds_b128 %0, %1, off offset:64\n\t"
        "global_load_async_to_lds_b128 %0, %1, off offset:80\n\t"
        "global_load_async_to_lds_b128 %0, %1, off offset:96\n\t"
        "global_load_async_to_lds_b128 %0, %1, off offset:112"
        :: "v"(lds), "v"(g) : "memory");
}

__device__ __forceinline__ void wait_async0() {
    asm volatile("s_wait_asynccnt 0x0" ::: "memory");
}

// ---------------- WMMA GEMM ----------------
// grid = (OUT/BN, M/BM) = (64, 64), block = 256 threads = 8 wave32.

__global__ void __launch_bounds__(256)
qlin_wmma_gemm(const _Float16* __restrict__ A,      // [M_TOTAL][IN_DIM] f16
               const _Float16* __restrict__ W,      // [OUT_DIM][IN_DIM] f16 (int-valued)
               const float*    __restrict__ scales, // [OUT_DIM]
               const float*    __restrict__ bias,   // [OUT_DIM]
               float*          __restrict__ C)      // [M_TOTAL][OUT_DIM]
{
    extern __shared__ __align__(16) char smem[];    // [2][A tile | B tile]

    const int tid  = threadIdx.x;
    const int wid  = tid >> 5;       // 0..7
    const int lane = tid & 31;
    const int hlf  = lane >> 4;      // lane half (ISA 7.12.2 layouts)
    const int l16  = lane & 15;

    const int bm = blockIdx.y;       // 0..63
    const int bn = blockIdx.x;       // 0..63
    const int wm = wid >> 2;         // 0..1  -> 64-row slab
    const int wn = wid & 3;          // 0..3  -> 64-col slab

    v8f acc[4][4];
#pragma unroll
    for (int mi = 0; mi < 4; ++mi)
#pragma unroll
        for (int ni = 0; ni < 4; ++ni)
            acc[mi][ni] = (v8f){};

    // staging map: A tile 128x64 halfs -> 2 threads/row x 64B; B tile 256x64 -> 1 thread/row x 128B
    const int arow = tid >> 1;
    const int acol = (tid & 1) * 32;                       // halfs
    const _Float16* gA = A + (size_t)(bm * BM + arow) * IN_DIM + acol;
    const _Float16* gB = W + (size_t)(bn * BN + tid) * IN_DIM;
    const unsigned lA_off = (unsigned)((arow * LDT + acol) * 2);
    const unsigned lB_off = (unsigned)(A_BYTES + tid * LDT * 2);
    const unsigned lds_base = lds_addr(smem);

    // prologue: fill buffer 0
    async_copy_64B (lds_base + lA_off, gA);
    async_copy_128B(lds_base + lB_off, gB);
    wait_async0();
    __syncthreads();

    for (int k0 = 0, it = 0; k0 < IN_DIM; k0 += BK, ++it) {
        const int cur = it & 1;
        const unsigned bufc = lds_base + (unsigned)(cur * BUF_BYTES);
        const unsigned bufn = lds_base + (unsigned)((cur ^ 1) * BUF_BYTES);

        if (k0 + BK < IN_DIM) {                  // issue next tile into other buffer
            async_copy_64B (bufn + lA_off, gA + k0 + BK);
            async_copy_128B(bufn + lB_off, gB + k0 + BK);
        }

        const _Float16* sA = (const _Float16*)(smem + cur * BUF_BYTES);
        const _Float16* sB = (const _Float16*)(smem + cur * BUF_BYTES + A_BYTES);

#pragma unroll
        for (int kk = 0; kk < 2; ++kk) {         // two K=32 WMMA slabs per BK=64 tile
            const int kb = kk * 32;

            // A fragment (16x32 f16): lane m=l16, K in [hlf*8,+8) U [16+hlf*8,+8)
            F16Frag af[4];
#pragma unroll
            for (int mi = 0; mi < 4; ++mi) {
                const _Float16* pa = sA + (wm * 64 + mi * 16 + l16) * LDT + kb;
                af[mi].h[0] = *(const v8h*)(pa + hlf * 8);
                af[mi].h[1] = *(const v8h*)(pa + 16 + hlf * 8);
            }
            // B fragment (32x16 f16): lane n=l16, K in [hlf*8,+8) U [16+hlf*8,+8)
            F16Frag bf[4];
#pragma unroll
            for (int ni = 0; ni < 4; ++ni) {
                const _Float16* pb = sB + (wn * 64 + ni * 16 + l16) * LDT + kb;
                bf[ni].h[0] = *(const v8h*)(pb + hlf * 8);
                bf[ni].h[1] = *(const v8h*)(pb + 16 + hlf * 8);
            }
#pragma unroll
            for (int mi = 0; mi < 4; ++mi)
#pragma unroll
                for (int ni = 0; ni < 4; ++ni)
                    acc[mi][ni] = __builtin_amdgcn_wmma_f32_16x16x32_f16(
                        false, af[mi].v, false, bf[ni].v,
                        (short)0, acc[mi][ni], false, false);
        }

        wait_async0();                           // my writes into bufn landed
        __syncthreads();                         // everyone done reading bufc / writing bufn
    }

    // epilogue: acc * scale[n] + bias[n] in f32 (exact weight-scale math)
#pragma unroll
    for (int ni = 0; ni < 4; ++ni) {
        const int n = bn * BN + wn * 64 + ni * 16 + l16;
        const float sc = scales[n];
        const float bi = bias[n];
#pragma unroll
        for (int mi = 0; mi < 4; ++mi) {
            const int mbase = bm * BM + wm * 64 + mi * 16 + hlf * 8;  // C: VGPR r -> M=r+8*hlf
#pragma unroll
            for (int r = 0; r < 8; ++r)
                C[(size_t)(mbase + r) * OUT_DIM + n] = acc[mi][ni][r] * sc + bi;
        }
    }
}

// ---------------- launch ----------------

extern "C" void kernel_launch(void* const* d_in, const int* in_sizes, int n_in,
                              void* d_out, int out_size, void* d_ws, size_t ws_size,
                              hipStream_t stream) {
    const float* x  = (const float*)d_in[0];   // [4,2048,4096] f32
    const int*   wq = (const int*)d_in[1];     // [16384,4096] int32 (int8 values)
    const float* sc = (const float*)d_in[2];   // [16384]
    const float* bi = (const float*)d_in[3];   // [16384]
    float* out = (float*)d_out;                // [4,2048,16384] f32

    _Float16* xh = (_Float16*)d_ws;                                          // 64 MB
    _Float16* wh = (_Float16*)((char*)d_ws +
                               (size_t)M_TOTAL * IN_DIM * sizeof(_Float16)); // +64 MB

    qlin_cvt_x_f16<<<(M_TOTAL * (size_t)IN_DIM) / (8 * 256), 256, 0, stream>>>(x, xh);
    qlin_cvt_w_f16<<<(OUT_DIM * (size_t)IN_DIM) / (8 * 256), 256, 0, stream>>>(wq, wh);

    // allow >64KB dynamic LDS (WGP has 320KB); immediate call, not a stream op
    (void)hipFuncSetAttribute((const void*)qlin_wmma_gemm,
                              hipFuncAttributeMaxDynamicSharedMemorySize, LDS_TOTAL);

    dim3 grid(OUT_DIM / BN, M_TOTAL / BM);     // 64 x 64 = 4096 workgroups
    qlin_wmma_gemm<<<grid, 256, LDS_TOTAL, stream>>>(xh, wh, sc, bi, out);
}
